// LocalGlobalAttentionGNN_60370060313265
// MI455X (gfx1250) — matrix operand order repaired and checked by reading, not compiled, not run on hardware
//
#include <hip/hip_runtime.h>
#include <hip/hip_bf16.h>
#include <math.h>

// ---------------------------------------------------------------------------
// Problem constants (match reference setup_inputs)
// ---------------------------------------------------------------------------
#define NN      16384        // nodes
#define GG      16           // graphs
#define NPG     1024         // nodes per graph
#define HH      128          // hidden
#define IN_C    32
#define E_EDGES 131072       // random edges (8*N)
#define EPE     147456       // edges + self loops
#define GHEADS  4
#define HD      32           // HH / GHEADS
#define kEPS    1e-5f

typedef __attribute__((ext_vector_type(16))) __bf16 v16bf;
typedef __attribute__((ext_vector_type(8)))  float  v8f;

#define DEV __device__ __forceinline__

DEV __bf16 to_bf(float x) { return (__bf16)x; }

// ordered-uint encoding for float atomicMax (monotone in f)
DEV unsigned enc_f(float f) {
  unsigned u = __float_as_uint(f);
  return (u & 0x80000000u) ? ~u : (u | 0x80000000u);
}
DEV float dec_f(unsigned e) {
  unsigned u = (e & 0x80000000u) ? (e ^ 0x80000000u) : ~e;
  return __uint_as_float(u);
}

// ---------------------------------------------------------------------------
// Generic WMMA GEMM:  C[M x Nc] = act(A[M x K] * W[K x Nc] + bias + R)
// BM=128, BN=64, BK=32; 8 waves (256 threads); per wave one 16-row strip.
// ACT: 0 none, 1 relu, 2 exact gelu, 3 sigmoid
// ---------------------------------------------------------------------------
template <int ACT>
__global__ __launch_bounds__(256)
void gemm_wmma(const float* __restrict__ A, const float* __restrict__ W,
               const float* __restrict__ bias, const float* __restrict__ R,
               float* __restrict__ C, int M, int K, int Nc)
{
  __shared__ __bf16 As[128 * 32];
  __shared__ __bf16 Bs[32 * 64];

  const int tid  = threadIdx.x;
  const int wave = tid >> 5;
  const int lane = tid & 31;
  const int half = lane >> 4;
  const int l15  = lane & 15;
  const int m0   = blockIdx.x * 128;
  const int n0   = blockIdx.y * 64;

  const v8f vz = {};
  v8f acc[4];
  acc[0] = vz; acc[1] = vz; acc[2] = vz; acc[3] = vz;

  for (int k0 = 0; k0 < K; k0 += 32) {
    for (int i = tid; i < 128 * 32; i += 256) {
      int r = i >> 5, c = i & 31;
      As[i] = to_bf(A[(size_t)(m0 + r) * K + (k0 + c)]);
    }
    for (int i = tid; i < 32 * 64; i += 256) {
      int r = i >> 6, c = i & 63;
      Bs[i] = to_bf(W[(size_t)(k0 + r) * Nc + (n0 + c)]);
    }
    __syncthreads();

    // A fragment: 16x32 strip, lane = row, K contiguous (k = half*16 + j)
    v16bf a;
    {
      int row = wave * 16 + l15;
#pragma unroll
      for (int j = 0; j < 16; ++j) {
        int kk = half * 16 + j;
        a[j] = As[row * 32 + kk];
      }
    }
#pragma unroll
    for (int t = 0; t < 4; ++t) {
      v16bf b;
      int col = t * 16 + l15;
#pragma unroll
      for (int j = 0; j < 16; ++j)
        b[j] = Bs[(half * 16 + j) * 64 + col];
      acc[t] = __builtin_amdgcn_wmma_f32_16x16x32_bf16(
          false, a, false, b, (short)0, acc[t], false, false);
    }
    __syncthreads();
  }

  // epilogue: C row = i + 8*half, col = lane&15 (per 16x16 tile)
#pragma unroll
  for (int t = 0; t < 4; ++t) {
    int n = n0 + t * 16 + l15;
    float bv = bias ? bias[n] : 0.0f;
#pragma unroll
    for (int i = 0; i < 8; ++i) {
      int m = m0 + wave * 16 + i + 8 * half;
      float v = acc[t][i] + bv;
      if (R) v += R[(size_t)m * Nc + n];
      if (ACT == 1)      v = fmaxf(v, 0.0f);
      else if (ACT == 2) v = 0.5f * v * (1.0f + erff(v * 0.70710678118f));
      else if (ACT == 3) v = 1.0f / (1.0f + expf(-v));
      C[(size_t)m * Nc + n] = v;
    }
  }
}

// ---------------------------------------------------------------------------
// Flash attention per (graph, head): Q tile 64 rows (4 waves), K/V tiles 128.
// Online softmax; QK^T and P*V on WMMA bf16. scale = 1/sqrt(HD).
// ---------------------------------------------------------------------------
__global__ __launch_bounds__(128)
void flash_attn(const float* __restrict__ q, const float* __restrict__ k,
                const float* __restrict__ v, float* __restrict__ ao)
{
  __shared__ __bf16 Ks[128 * 32];
  __shared__ __bf16 Vs[128 * 32];
  __shared__ __bf16 Ps[4][16 * 128];

  const int tid  = threadIdx.x;
  const int wave = tid >> 5;
  const int lane = tid & 31;
  const int half = lane >> 4;
  const int l15  = lane & 15;

  const int bid = blockIdx.x;
  const int qt  = bid & 15;        // 16 q-tiles of 64 rows
  const int gh  = bid >> 4;
  const int h   = gh & (GHEADS - 1);
  const int g   = gh >> 2;
  const float scale = 0.17677669529663687f;   // 1/sqrt(32)

  const int qrow0 = qt * 64 + wave * 16;      // base row (within graph) of this wave

  // Q fragment loaded once straight from global (A layout)
  v16bf qa;
  {
    const float* qp = q + (size_t)(g * NPG + qrow0 + l15) * HH + h * HD;
#pragma unroll
    for (int j = 0; j < 16; ++j) {
      int d = half * 16 + j;
      qa[j] = to_bf(qp[d]);
    }
  }

  const v8f vz = {};
  v8f oacc[2]; oacc[0] = vz; oacc[1] = vz;
  float mrow[8], lrow[8];
#pragma unroll
  for (int i = 0; i < 8; ++i) { mrow[i] = -3.0e38f; lrow[i] = 0.0f; }

  for (int kt = 0; kt < NPG / 128; ++kt) {
    for (int i = tid; i < 128 * 32; i += 128) {
      int r = i >> 5, c = i & 31;
      size_t base = (size_t)(g * NPG + kt * 128 + r) * HH + h * HD + c;
      Ks[i] = to_bf(k[base]);
      Vs[i] = to_bf(v[base]);
    }
    __syncthreads();

    // S = Q * K^T : 8 column tiles of 16 keys
    v8f s[8];
#pragma unroll
    for (int t = 0; t < 8; ++t) {
      v16bf b;
      int col = t * 16 + l15;           // key index within tile
#pragma unroll
      for (int j = 0; j < 16; ++j)
        b[j] = Ks[col * 32 + half * 16 + j];   // B[d][key]: d = half*16+j
      s[t] = __builtin_amdgcn_wmma_f32_16x16x32_bf16(
          false, qa, false, b, (short)0, vz, false, false);
    }

    // online softmax per row (row = i + 8*half; 16 cols per lane-group)
    float corr[8];
#pragma unroll
    for (int i = 0; i < 8; ++i) {
      float rm = -3.0e38f;
#pragma unroll
      for (int t = 0; t < 8; ++t) rm = fmaxf(rm, s[t][i]);
      rm *= scale;
      rm = fmaxf(rm, __shfl_xor(rm, 1, 32));
      rm = fmaxf(rm, __shfl_xor(rm, 2, 32));
      rm = fmaxf(rm, __shfl_xor(rm, 4, 32));
      rm = fmaxf(rm, __shfl_xor(rm, 8, 32));
      float mn = fmaxf(mrow[i], rm);
      corr[i] = expf(mrow[i] - mn);
      mrow[i] = mn;
      float rs = 0.0f;
#pragma unroll
      for (int t = 0; t < 8; ++t) {
        float p = expf(s[t][i] * scale - mn);
        s[t][i] = p;
        rs += p;
      }
      rs += __shfl_xor(rs, 1, 32);
      rs += __shfl_xor(rs, 2, 32);
      rs += __shfl_xor(rs, 4, 32);
      rs += __shfl_xor(rs, 8, 32);
      lrow[i] = lrow[i] * corr[i] + rs;
      oacc[0][i] *= corr[i];
      oacc[1][i] *= corr[i];
    }

    // P strip -> LDS (row-major 16x128 per wave), then re-load as A fragments
#pragma unroll
    for (int t = 0; t < 8; ++t)
#pragma unroll
      for (int i = 0; i < 8; ++i)
        Ps[wave][(i + 8 * half) * 128 + t * 16 + l15] = to_bf(s[t][i]);

#pragma unroll
    for (int kb = 0; kb < 4; ++kb) {
      v16bf pa;
#pragma unroll
      for (int j = 0; j < 16; ++j) {
        int kk = half * 16 + j;
        pa[j] = Ps[wave][l15 * 128 + kb * 32 + kk];
      }
#pragma unroll
      for (int c = 0; c < 2; ++c) {
        v16bf b;
#pragma unroll
        for (int j = 0; j < 16; ++j)
          b[j] = Vs[(kb * 32 + half * 16 + j) * 32 + c * 16 + l15];
        oacc[c] = __builtin_amdgcn_wmma_f32_16x16x32_bf16(
            false, pa, false, b, (short)0, oacc[c], false, false);
      }
    }
    __syncthreads();
  }

#pragma unroll
  for (int c = 0; c < 2; ++c)
#pragma unroll
    for (int i = 0; i < 8; ++i) {
      int n = g * NPG + qrow0 + i + 8 * half;
      int d = c * 16 + l15;
      ao[(size_t)n * HH + h * HD + d] = oacc[c][i] / lrow[i];
    }
}

// ---------------------------------------------------------------------------
// LayerNorm over H=128: one wave per row (8 waves / block)
// ---------------------------------------------------------------------------
__global__ __launch_bounds__(256)
void layernorm_k(const float* __restrict__ x, const float* __restrict__ g,
                 const float* __restrict__ b, float* __restrict__ y)
{
  int wave = threadIdx.x >> 5, lane = threadIdx.x & 31;
  int row = blockIdx.x * 8 + wave;
  if (row >= NN) return;
  const float* xr = x + (size_t)row * HH;
  float vals[4];
  float s = 0.0f;
#pragma unroll
  for (int j = 0; j < 4; ++j) { vals[j] = xr[lane + 32 * j]; s += vals[j]; }
  for (int m = 1; m < 32; m <<= 1) s += __shfl_xor(s, m, 32);
  float mean = s * (1.0f / HH);
  float vs = 0.0f;
#pragma unroll
  for (int j = 0; j < 4; ++j) { float d = vals[j] - mean; vs += d * d; }
  for (int m = 1; m < 32; m <<= 1) vs += __shfl_xor(vs, m, 32);
  float rstd = rsqrtf(vs * (1.0f / HH) + kEPS);
#pragma unroll
  for (int j = 0; j < 4; ++j) {
    int c = lane + 32 * j;
    y[(size_t)row * HH + c] = (vals[j] - mean) * rstd * g[c] + b[c];
  }
}

// ---------------------------------------------------------------------------
// GAT helpers
// ---------------------------------------------------------------------------
// per-node, per-head attention logits: wave per (n, h)
__global__ __launch_bounds__(256)
void gat_alpha(const float* __restrict__ xh, const float* __restrict__ att_src,
               const float* __restrict__ att_dst, float* __restrict__ asrc,
               float* __restrict__ adst, int heads)
{
  int wave = threadIdx.x >> 5, lane = threadIdx.x & 31;
  int id = blockIdx.x * 8 + wave;
  if (id >= NN * heads) return;
  int n = id / heads, h = id - n * heads;
  const float* xp = xh + (size_t)n * heads * HH + h * HH;
  float s1 = 0.0f, s2 = 0.0f;
#pragma unroll
  for (int j = 0; j < 4; ++j) {
    int c = lane + 32 * j;
    float xv = xp[c];
    s1 += xv * att_src[h * HH + c];
    s2 += xv * att_dst[h * HH + c];
  }
  for (int m = 1; m < 32; m <<= 1) {
    s1 += __shfl_xor(s1, m, 32);
    s2 += __shfl_xor(s2, m, 32);
  }
  if (lane == 0) { asrc[id] = s1; adst[id] = s2; }
}

// pass 1: leaky_relu logit per edge, segment-max via ordered-uint atomicMax
__global__ void gat_edge1(const int* __restrict__ ei, const float* __restrict__ asrc,
                          const float* __restrict__ adst, float* __restrict__ ea,
                          unsigned* __restrict__ segenc, int heads)
{
  int id = blockIdx.x * 256 + threadIdx.x;
  if (id >= EPE * heads) return;
  int e = id / heads, h = id - e * heads;
  int s, d;
  if (e < E_EDGES) { s = ei[e]; d = ei[E_EDGES + e]; }
  else             { s = d = e - E_EDGES; }        // self loop
  float a = asrc[s * heads + h] + adst[d * heads + h];
  a = (a > 0.0f) ? a : 0.2f * a;                   // leaky_relu 0.2
  ea[id] = a;
  atomicMax(&segenc[d * heads + h], enc_f(a));
}

// pass 2: exp & segment-sum denominator
__global__ void gat_edge2(const int* __restrict__ ei, const float* __restrict__ ea,
                          const unsigned* __restrict__ segenc, float* __restrict__ eex,
                          float* __restrict__ denom, int heads)
{
  int id = blockIdx.x * 256 + threadIdx.x;
  if (id >= EPE * heads) return;
  int e = id / heads, h = id - e * heads;
  int d = (e < E_EDGES) ? ei[E_EDGES + e] : (e - E_EDGES);
  float ex = expf(ea[id] - dec_f(segenc[d * heads + h]));
  eex[id] = ex;
  atomicAdd(&denom[d * heads + h], ex);
}

// pass 3: weighted scatter of xh[src] into out[dst]; wave per (edge, head)
__global__ __launch_bounds__(256)
void gat_edge3(const int* __restrict__ ei, const float* __restrict__ xh,
               const float* __restrict__ eex, const float* __restrict__ denom,
               float* __restrict__ out, int heads)
{
  int wave = threadIdx.x >> 5, lane = threadIdx.x & 31;
  int id = blockIdx.x * 8 + wave;
  if (id >= EPE * heads) return;
  int e = id / heads, h = id - e * heads;
  int s, d;
  if (e < E_EDGES) { s = ei[e]; d = ei[E_EDGES + e]; }
  else             { s = d = e - E_EDGES; }
  float coef = eex[id] / (denom[d * heads + h] + 1e-16f);
  const float* xp = xh + (size_t)s * heads * HH + h * HH;
  float* op = out + (size_t)d * heads * HH + h * HH;
#pragma unroll
  for (int j = 0; j < 4; ++j) {
    int c = lane + 32 * j;
    atomicAdd(&op[c], coef * xp[c]);
  }
}

// bn(eval) + relu (+ elementwise skip when the skip projection is identity)
__global__ void bn_relu_skip(const float* __restrict__ x, const float* __restrict__ g,
                             const float* __restrict__ b, const float* __restrict__ m,
                             const float* __restrict__ vv, const float* __restrict__ skip,
                             float* __restrict__ y, int C)
{
  int id = blockIdx.x * 256 + threadIdx.x;
  if (id >= NN * C) return;
  int c = id % C;
  float val = (x[id] - m[c]) * rsqrtf(vv[c] + kEPS) * g[c] + b[c];
  val = fmaxf(val, 0.0f);
  if (skip) val += skip[id];
  y[id] = val;
}

// cat = [h_local | h_global]
__global__ void concat_k(const float* __restrict__ hl, const float* __restrict__ hg,
                         float* __restrict__ cat)
{
  int id = blockIdx.x * 256 + threadIdx.x;
  if (id >= NN * 256) return;
  int n = id >> 8, c = id & 255;
  cat[id] = (c < HH) ? hl[(size_t)n * HH + c] : hg[(size_t)n * HH + (c - HH)];
}

// fused = gate*hl + (1-gate)*hg
__global__ void fuse_k(const float* __restrict__ gate, const float* __restrict__ hl,
                       const float* __restrict__ hg, float* __restrict__ fused)
{
  int id = blockIdx.x * 256 + threadIdx.x;
  if (id >= NN * HH) return;
  float gv = gate[id];
  fused[id] = gv * hl[id] + (1.0f - gv) * hg[id];
}

// tiny head: [N x 64] x [64 x 2]
__global__ void head2_k(const float* __restrict__ hid, const float* __restrict__ W,
                        const float* __restrict__ b, float* __restrict__ out)
{
  int n = blockIdx.x * 256 + threadIdx.x;
  if (n >= NN) return;
  float s0 = b[0], s1 = b[1];
#pragma unroll 8
  for (int k = 0; k < 64; ++k) {
    float hv = hid[(size_t)n * 64 + k];
    s0 += hv * W[k * 2 + 0];
    s1 += hv * W[k * 2 + 1];
  }
  out[n * 2 + 0] = s0;
  out[n * 2 + 1] = s1;
}

// ---------------------------------------------------------------------------
// Host orchestration
// ---------------------------------------------------------------------------
static void gemm(const float* A, const float* W, const float* bias, const float* R,
                 float* C, int M, int K, int Nc, int act, hipStream_t s)
{
  dim3 g(M / 128, Nc / 64), blk(256);
  switch (act) {
    case 1:  gemm_wmma<1><<<g, blk, 0, s>>>(A, W, bias, R, C, M, K, Nc); break;
    case 2:  gemm_wmma<2><<<g, blk, 0, s>>>(A, W, bias, R, C, M, K, Nc); break;
    case 3:  gemm_wmma<3><<<g, blk, 0, s>>>(A, W, bias, R, C, M, K, Nc); break;
    default: gemm_wmma<0><<<g, blk, 0, s>>>(A, W, bias, R, C, M, K, Nc); break;
  }
}

extern "C" void kernel_launch(void* const* d_in, const int* in_sizes, int n_in,
                              void* d_out, int out_size, void* d_ws, size_t ws_size,
                              hipStream_t stream)
{
  (void)in_sizes; (void)n_in; (void)out_size; (void)ws_size;

  const float* x  = (const float*)d_in[0];
  const int*   ei = (const int*)d_in[1];
  // d_in[2] = batch (unused: graphs are block-contiguous and equal size)
  const float* P[64];
  for (int i = 3; i < 63; ++i) P[i] = (const float*)d_in[i];

  // --- workspace carve-up (floats) ---
  float* ws   = (float*)d_ws;
  const size_t NH = (size_t)NN * HH;          // N*128
  const size_t N5 = (size_t)NN * 512;
  float* h     = ws;                          // N*128, persists
  float* bufA  = h + NH;                      // N*512  (xh; later q|k|v, ffn_mid, cat)
  float* bufB  = bufA + N5;                   // N*512  (gatout; later ao|x1|h1|x2)
  float* bufC  = bufB + N5;                   // N*512  (h_local ping; later hg|gate|fused|hid)
  float* bufD  = bufC + N5;                   // N*512  (h_local pong; final h_local)
  float* asrc  = bufD + N5;                   // N*4
  float* adst  = asrc + (size_t)NN * 4;       // N*4
  unsigned* segenc = (unsigned*)(adst + (size_t)NN * 4); // N*4
  float* denom = (float*)segenc + (size_t)NN * 4;        // N*4
  float* ea    = denom + (size_t)NN * 4;      // EPE*4
  float* eex   = ea + (size_t)EPE * 4;        // EPE*4

  // ---- input linear: h = x @ W_in + b_in ----
  gemm(x, P[3], P[4], nullptr, h, NN, IN_C, HH, 0, stream);

  // ---- 4 GAT layers ----
  struct L { int W, as, ad, bng, bnb, bnm, bnv, skW, skb;
             int in_dim, heads, out_dim; };
  const L layers[4] = {
    { 5,  6,  7,  9, 10, 11, 12, 13, 14, 128, 4, 512},
    {15, 16, 17, 19, 20, 21, 22, -1, -1, 512, 4, 512},
    {23, 24, 25, 27, 28, 29, 30, -1, -1, 512, 4, 512},
    {31, 32, 33, 35, 36, 37, 38, 39, 40, 512, 1, 128},
  };
  float* lin[4]  = {h,    bufC, bufD, bufC};
  float* lout[4] = {bufC, bufD, bufC, bufD};

  for (int li = 0; li < 4; ++li) {
    const L& l = layers[li];
    const int heads = l.heads;
    const int hhH   = heads * HH;             // == out_dim for all layers here

    // xh = in @ W (no bias in GATConv linear)
    gemm(lin[li], P[l.W], nullptr, nullptr, bufA, NN, l.in_dim, hhH, 0, stream);

    // per-node attention logits
    {
      int waves = NN * heads;
      gat_alpha<<<(waves + 7) / 8, 256, 0, stream>>>(bufA, P[l.as], P[l.ad],
                                                     asrc, adst, heads);
    }
    // zero segment state + output accumulator
    hipMemsetAsync(segenc, 0, (size_t)NN * heads * sizeof(unsigned), stream);
    hipMemsetAsync(denom,  0, (size_t)NN * heads * sizeof(float), stream);
    hipMemsetAsync(bufB,   0, (size_t)NN * hhH * sizeof(float), stream);

    int tot = EPE * heads;
    gat_edge1<<<(tot + 255) / 256, 256, 0, stream>>>(ei, asrc, adst, ea, segenc, heads);
    gat_edge2<<<(tot + 255) / 256, 256, 0, stream>>>(ei, ea, segenc, eex, denom, heads);
    gat_edge3<<<(tot + 7) / 8, 256, 0, stream>>>(ei, bufA, eex, denom, bufB, heads);

    // bn + relu (+ identity skip when no projection)
    const float* idskip = (l.skW < 0) ? lin[li] : nullptr;
    bn_relu_skip<<<((size_t)NN * hhH + 255) / 256, 256, 0, stream>>>(
        bufB, P[l.bng], P[l.bnb], P[l.bnm], P[l.bnv], idskip, lout[li], hhH);

    // projected skip: out += in @ skip.W + skip.b  (residual path of GEMM)
    if (l.skW >= 0)
      gemm(lin[li], P[l.skW], P[l.skb], lout[li], lout[li],
           NN, l.in_dim, l.out_dim, 0, stream);
  }
  float* h_local = bufD;   // final GAT output [N x 128]

  // ---- global branch ----
  float* qb = bufA;            // reuse xh space
  float* kb = bufA + NH;
  float* vb = bufA + 2 * NH;
  gemm(h, P[41], P[42], nullptr, qb, NN, HH, HH, 0, stream);
  gemm(h, P[43], P[44], nullptr, kb, NN, HH, HH, 0, stream);
  gemm(h, P[45], P[46], nullptr, vb, NN, HH, HH, 0, stream);

  float* ao = bufB;
  flash_attn<<<GG * GHEADS * (NPG / 64), 128, 0, stream>>>(qb, kb, vb, ao);

  float* x1 = bufB + NH;
  gemm(ao, P[47], P[48], /*R=*/h, x1, NN, HH, HH, 0, stream);   // o-proj + residual
  float* h1 = bufB + 2 * NH;
  layernorm_k<<<NN / 8, 256, 0, stream>>>(x1, P[49], P[50], h1);

  float* ffn_mid = bufA;                                         // q/k/v now dead
  gemm(h1, P[51], P[52], nullptr, ffn_mid, NN, HH, 2 * HH, 2, stream);  // GELU
  float* x2 = bufB + 3 * NH;
  gemm(ffn_mid, P[53], P[54], /*R=*/h1, x2, NN, 2 * HH, HH, 0, stream);
  float* h_global = bufC;                                        // bufC free after L3
  layernorm_k<<<NN / 8, 256, 0, stream>>>(x2, P[55], P[56], h_global);

  // ---- gated fusion + head ----
  float* cat = bufA;                                             // ffn_mid dead
  concat_k<<<((size_t)NN * 256 + 255) / 256, 256, 0, stream>>>(h_local, h_global, cat);
  float* gate = bufC + NH;
  gemm(cat, P[57], P[58], nullptr, gate, NN, 2 * HH, HH, 3, stream);   // sigmoid
  float* fused = bufC + 2 * NH;
  fuse_k<<<(NH + 255) / 256, 256, 0, stream>>>(gate, h_local, h_global, fused);
  float* hid = bufC + 3 * NH;
  gemm(fused, P[59], P[60], nullptr, hid, NN, HH, 64, 1, stream);      // relu
  head2_k<<<(NN + 255) / 256, 256, 0, stream>>>(hid, P[61], P[62], (float*)d_out);
}
